// Generator_6373731467428
// MI455X (gfx1250) — matrix operand order repaired and checked by reading, not compiled, not run on hardware
//
#include <hip/hip_runtime.h>
#include <stdint.h>

// ---------------------------------------------------------------------------
// CDNA5 (gfx1250) UGATIT-style generator forward pass.
// Conv/attention GEMMs -> v_wmma_f32_16x16x32_bf16 (wave32).
// A-tiles staged by the Tensor Data Mover (tensor_load_to_lds),
// B-tiles gathered with global_load_async_to_lds_b32,
// double-buffered and software-pipelined on ASYNCcnt/TENSORcnt.
// ---------------------------------------------------------------------------

typedef __bf16 bf16;
typedef __attribute__((ext_vector_type(16))) __bf16 v16bf;
typedef __attribute__((ext_vector_type(8)))  float  v8f;
typedef __attribute__((ext_vector_type(4)))  unsigned int v4u;
typedef __attribute__((ext_vector_type(8)))  unsigned int v8u;

#if defined(__has_builtin)
#if __has_builtin(__builtin_amdgcn_s_wait_asynccnt)
#define WAIT_ASYNC(n) __builtin_amdgcn_s_wait_asynccnt(n)
#endif
#if __has_builtin(__builtin_amdgcn_s_wait_tensorcnt)
#define WAIT_TENSOR(n) __builtin_amdgcn_s_wait_tensorcnt(n)
#endif
#endif
#ifndef WAIT_ASYNC
#define WAIT_ASYNC(n) do { if ((n) == 0) asm volatile("s_wait_asynccnt 0x0" ::: "memory"); \
                           else          asm volatile("s_wait_asynccnt 0x8" ::: "memory"); } while (0)
#endif
#ifndef WAIT_TENSOR
#define WAIT_TENSOR(n) do { if ((n) == 0) asm volatile("s_wait_tensorcnt 0x0" ::: "memory"); \
                            else          asm volatile("s_wait_tensorcnt 0x1" ::: "memory"); } while (0)
#endif

__device__ __forceinline__ bf16 f2bf(float f) {
  union { float f; uint32_t u; } v; v.f = f;
  uint32_t r = (v.u + 0x7FFFu + ((v.u >> 16) & 1u)) >> 16;
  union { uint16_t u; bf16 h; } o; o.u = (uint16_t)r;
  return o.h;
}

// ---------------------------------------------------------------------------
// Zero-fill (for the async-gather OOB landing pad; d_ws is poisoned by harness)
// ---------------------------------------------------------------------------
__global__ void zero_fill(float* __restrict__ p, int n) {
  int i = blockIdx.x * 256 + threadIdx.x;
  if (i < n) p[i] = 0.f;
}

// ---------------------------------------------------------------------------
// Weight packing: f32 conv weights -> zero-padded bf16 [Mpad][Kpad] row-major.
// transposed=1: ConvTranspose transform Wt[co][ci,ky,kx] = w[ci,co,kh-1-ky,kw-1-kx]
// ---------------------------------------------------------------------------
__global__ void pack_weights(const float* __restrict__ w, bf16* __restrict__ out,
                             int Cout, int Cin, int kh, int kw,
                             int Kdim, int Mpad, int Kpad, int transposed) {
  int i = blockIdx.x * 256 + threadIdx.x;
  int total = Mpad * Kpad;
  if (i >= total) return;
  int m = i / Kpad, k = i % Kpad;
  float v = 0.f;
  if (m < Cout && k < Kdim) {
    int khkw = kh * kw;
    int ci = k / khkw, r = k % khkw;
    int ky = r / kw, kx = r % kw;
    if (!transposed)
      v = w[(((size_t)m * Cin + ci) * kh + ky) * kw + kx];
    else
      v = w[(((size_t)ci * Cout + m) * kh + (kh - 1 - ky)) * kw + (kw - 1 - kx)];
  }
  out[i] = f2bf(v);
}

// ---------------------------------------------------------------------------
// Implicit-GEMM conv via WMMA bf16 (M=Cout, N=B*Ho*Wo, K=Cin*kh*kw).
// Block tile 64x64, BK=32, 256 threads = 8 waves (4 M x 2 N),
// each wave: 16x32 output sub-tile = two 16x16 WMMA accumulators.
// Double-buffered LDS; tile t+1 streamed by TDM/async while tile t computes.
// mode==0: normal conv. mode==1: ConvTranspose k3,s2,p1,op1. act==1: tanh.
// ---------------------------------------------------------------------------
#define BM 64
#define BN 64
#define BK 32

__global__ __launch_bounds__(256) void conv_gemm(
    const bf16* __restrict__ Wp, const float* __restrict__ X,
    const float* __restrict__ bias, const float* __restrict__ Zpad,
    float* __restrict__ Y,
    int N, int Cin, int H, int W, int Cout, int Hout, int Wout,
    int kh, int kw, int stride, int pad, int mode, int act,
    int Kdim, int Kpad, int Npos) {
  __shared__ bf16  sA[2][BM][BK];   // weight tiles (bf16, TDM-loaded)
  __shared__ float sB[2][BN][BK];   // patch tiles  (f32, async-gathered)

  const int tid  = threadIdx.x;
  const int lane = tid & 31;
  const int wave = tid >> 5;       // 0..7
  const int wm   = wave >> 1;      // 0..3  (M sub-tile)
  const int wn   = wave & 1;       // 0..1  (N sub-tile, 32 cols)
  const int m0   = blockIdx.y * BM;
  const int n0   = blockIdx.x * BN;

  v8f acc0 = {0.f,0.f,0.f,0.f,0.f,0.f,0.f,0.f};
  v8f acc1 = {0.f,0.f,0.f,0.f,0.f,0.f,0.f,0.f};

  // --- B-tile gather mapping: thread covers 1 column x 8 K values ---
  const int bcol = tid & 63;
  const int bk0  = (tid >> 6) * 8;
  int pos = n0 + bcol;
  const bool validpos = pos < Npos;
  int pp  = validpos ? pos : 0;
  const int hw  = Hout * Wout;
  const int nb  = pp / hw;
  const int rem = pp % hw;
  const int oy  = rem / Wout, ox = rem % Wout;
  const int khkw = kh * kw;
  const float* Xn = X + (size_t)nb * Cin * H * W;

  const int half = lane >> 4;       // 0/1
  const int l15  = lane & 15;
  const int kbA  = half * 8;        // A fragment: K base per half-wave
  const int ksB  = half * 16;       // unused; B fragment also uses kbA

  // Issue all transfers for one K-tile into LDS buffer `buf`.
  // Every lane always issues exactly 8 async loads (OOB lanes hit the
  // zero-filled landing pad) so per-wave ASYNCcnt is deterministic (=8).
  auto stage = [&](int k0, int buf) {
    if (wave == 0) {
      // TDM D# group0: count=1, lds_addr, global_addr, type=2 ("image")
      //     D# group1: data_size=2B, tensor 32x64, tile 32x64, stride Kpad
      unsigned long long ga =
          (unsigned long long)(uintptr_t)(Wp + (size_t)m0 * Kpad + k0);
      unsigned lds_a = (unsigned)(uintptr_t)&sA[buf][0][0];
      v4u g0;
      g0[0] = 1u;
      g0[1] = lds_a;
      g0[2] = (unsigned)ga;
      g0[3] = (unsigned)(ga >> 32) | (2u << 30);
      v8u g1;
      g1[0] = 1u << 16;        // data_size = 2 bytes
      g1[1] = 32u << 16;       // tensor_dim0 = 32
      g1[2] = 64u << 16;       // tensor_dim1 = 64
      g1[3] = 32u << 16;       // tile_dim0 = 32
      g1[4] = 64u;             // tile_dim1 = 64
      g1[5] = (unsigned)Kpad;  // tensor_dim0_stride
      g1[6] = 0u;
      g1[7] = 0u;
      asm volatile("tensor_load_to_lds %0, %1" :: "s"(g0), "s"(g1) : "memory");
    }
    int kbase = k0 + bk0;
    int ci = kbase / khkw;
    int r  = kbase - ci * khkw;
    int ky = r / kw;
    int kx = r - ky * kw;
    #pragma unroll
    for (int j = 0; j < 8; ++j) {
      bool inK = validpos && (kbase + j < Kdim);
      int iy, ix;
      bool ok;
      if (mode == 0) {
        iy = oy * stride - pad + ky;
        ix = ox * stride - pad + kx;
        ok = inK & (iy >= 0) & (iy < H) & (ix >= 0) & (ix < W);
      } else {  // transposed conv: dilated input, pad (1,2)
        int ty = oy - 1 + ky;
        int tx = ox - 1 + kx;
        ok = inK & (ty >= 0) & (ty < 2 * H) & (!(ty & 1)) &
             (tx >= 0) & (tx < 2 * W) & (!(tx & 1));
        iy = ty >> 1; ix = tx >> 1;
      }
      const float* gp = ok ? (Xn + ((size_t)ci * H + iy) * W + ix) : Zpad;
      unsigned lds_b = (unsigned)(uintptr_t)&sB[buf][bcol][bk0 + j];
      asm volatile("global_load_async_to_lds_b32 %0, %1, off"
                   :: "v"(lds_b),
                      "v"((unsigned long long)(uintptr_t)gp)
                   : "memory");
      if (++kx == kw) { kx = 0; if (++ky == kh) { ky = 0; ++ci; } }
    }
  };

  const int nt = Kpad / BK;
  stage(0, 0);
  for (int t = 0; t < nt; ++t) {
    const int cur = t & 1;
    if (t + 1 < nt) {
      stage((t + 1) * BK, (t + 1) & 1);  // stream next tile into other buffer
      WAIT_ASYNC(8);                     // in-order: tile t's 8 loads done
      if (wave == 0) WAIT_TENSOR(1);     // tile t's TDM done
    } else {
      WAIT_ASYNC(0);
      if (wave == 0) WAIT_TENSOR(0);
    }
    __syncthreads();                     // tile t visible to all waves

    // ---- fragments per ISA 7.12.2 layouts + WMMA ------------------------
    v16bf af, bf0, bf1;
    const int amrow = wm * 16 + l15;
    const int c0    = wn * 32 + l15;
    #pragma unroll
    for (int rr = 0; rr < 8; ++rr) {
      int kidx = kbA + ((rr >= 4) ? 16 : 0) + 2 * (rr & 3);
      af[2 * rr]     = sA[cur][amrow][kidx];
      af[2 * rr + 1] = sA[cur][amrow][kidx + 1];
    }
    #pragma unroll
    for (int e = 0; e < 16; ++e) {
      bf0[e] = f2bf(sB[cur][c0][kbA + (e & 7) + ((e >> 3) << 4)]);
      bf1[e] = f2bf(sB[cur][c0 + 16][kbA + (e & 7) + ((e >> 3) << 4)]);
    }
    acc0 = __builtin_amdgcn_wmma_f32_16x16x32_bf16(false, af, false, bf0,
                                                   (short)0, acc0, false, false);
    acc1 = __builtin_amdgcn_wmma_f32_16x16x32_bf16(false, af, false, bf1,
                                                   (short)0, acc1, false, false);
    __syncthreads();                     // all reads of buffer `cur` done
  }

  // ---- epilogue: C layout -> VGPR r holds M = r + 8*half, N = lane&15 ----
  #pragma unroll
  for (int rr = 0; rr < 8; ++rr) {
    int m = m0 + wm * 16 + rr + half * 8;
    if (m < Cout) {
      float bv = bias ? bias[m] : 0.f;
      #pragma unroll
      for (int t = 0; t < 2; ++t) {
        int n = n0 + wn * 32 + t * 16 + l15;
        if (n < Npos) {
          int nb2 = n / hw;
          int r2  = n % hw;
          float v = (t == 0 ? acc0[rr] : acc1[rr]) + bv;
          if (act == 1) v = tanhf(v);
          Y[((size_t)nb2 * Cout + m) * (size_t)hw + r2] = v;
        }
      }
    }
  }
}

// ---------------------------------------------------------------------------
// Per-(n,c) sum / sum-of-squares over H*W.  grid = N*C blocks.
// ---------------------------------------------------------------------------
__global__ void stats_kernel(const float* __restrict__ x, float* __restrict__ st,
                             int HW) {
  __shared__ float s1[256], s2[256];
  size_t base = (size_t)blockIdx.x * HW;
  float a = 0.f, b = 0.f;
  for (int i = threadIdx.x; i < HW; i += 256) {
    float v = x[base + i];
    a += v; b += v * v;
  }
  s1[threadIdx.x] = a; s2[threadIdx.x] = b;
  __syncthreads();
  for (int s = 128; s > 0; s >>= 1) {
    if (threadIdx.x < s) {
      s1[threadIdx.x] += s1[threadIdx.x + s];
      s2[threadIdx.x] += s2[threadIdx.x + s];
    }
    __syncthreads();
  }
  if (threadIdx.x == 0) {
    st[2 * blockIdx.x]     = s1[0];
    st[2 * blockIdx.x + 1] = s2[0];
  }
}

__global__ void layer_reduce(const float* __restrict__ st, float* __restrict__ lst,
                             int N, int C, int HW) {
  int c = blockIdx.x * blockDim.x + threadIdx.x;
  if (c >= C) return;
  float a = 0.f, b = 0.f;
  for (int n = 0; n < N; ++n) {
    a += st[2 * (n * C + c)];
    b += st[2 * (n * C + c) + 1];
  }
  float inv = 1.f / ((float)N * (float)HW);
  float m = a * inv;
  lst[2 * c]     = m;
  lst[2 * c + 1] = b * inv - m * m;
}

__global__ void adalin_apply(float* __restrict__ x, const float* __restrict__ st,
                             const float* __restrict__ lst,
                             const float* __restrict__ rho,
                             const float* __restrict__ gam,
                             const float* __restrict__ bet,
                             int C, int HW, size_t total, int relu) {
  float invHW = 1.f / (float)HW;
  for (size_t i = (size_t)blockIdx.x * 256 + threadIdx.x; i < total;
       i += (size_t)gridDim.x * 256) {
    size_t nc = i / (size_t)HW;
    int c = (int)(nc % (size_t)C);
    float s1 = st[2 * nc], s2 = st[2 * nc + 1];
    float im = s1 * invHW, iv = s2 * invHW - im * im;
    float lm = lst[2 * c], lv = lst[2 * c + 1];
    float r  = rho[c];
    float mean = r * im + (1.f - r) * lm;
    float var  = r * iv + (1.f - r) * lv;
    float y = gam[c] * (x[i] - mean) * rsqrtf(var + 1e-5f) + bet[c];
    if (relu) y = fmaxf(y, 0.f);
    x[i] = y;
  }
}

__global__ void inorm_apply(const float* __restrict__ xin, float* __restrict__ xout,
                            const float* __restrict__ st,
                            const float* __restrict__ res,
                            int HW, size_t total, int relu) {
  float invHW = 1.f / (float)HW;
  for (size_t i = (size_t)blockIdx.x * 256 + threadIdx.x; i < total;
       i += (size_t)gridDim.x * 256) {
    size_t nc = i / (size_t)HW;
    float im = st[2 * nc] * invHW;
    float iv = st[2 * nc + 1] * invHW - im * im;
    float y = (xin[i] - im) * rsqrtf(iv + 1e-5f);
    if (res) y += res[i];
    if (relu) y = fmaxf(y, 0.f);
    xout[i] = y;
  }
}

// ---------------------------------------------------------------------------
// Windowed attention: one block per 8x8 window (stride 4).
// scores = f^T g (64x64, K=32): 1 WMMA/tile; op = h att^T (256x64, K=64): 2.
// ---------------------------------------------------------------------------
__global__ __launch_bounds__(256) void win_attn(
    const float* __restrict__ F, const float* __restrict__ G,
    const float* __restrict__ Hh, float* __restrict__ out,
    int B, int C, int Cf, int Hdim, int Wdim, int Nh, int Nw) {
  __shared__ bf16  fA[64][32];
  __shared__ bf16  gB[64][32];
  __shared__ bf16  hA[256][64];
  __shared__ float sc[64][65];
  __shared__ bf16  aB[64][64];
  __shared__ float kmask[64];

  const int b  = blockIdx.x / (Nh * Nw);
  const int rw = blockIdx.x % (Nh * Nw);
  const int wy = rw / Nw, wx = rw % Nw;
  const int tid = threadIdx.x;

  {
    int p  = tid & 63;
    int c0 = (tid >> 6) * 8;
    int py = p >> 3, px = p & 7;
    int gy = wy * 4 + py, gx = wx * 4 + px;
    bool ok = (gy < Hdim) && (gx < Wdim);
    if (tid < 64) kmask[tid] = ok ? 0.f : -1e30f;
    #pragma unroll
    for (int j = 0; j < 8; ++j) {
      int c = c0 + j;
      float fv = 0.f, gv = 0.f;
      if (ok) {
        size_t idx = (((size_t)b * Cf + c) * Hdim + gy) * Wdim + gx;
        fv = F[idx]; gv = G[idx];
      }
      fA[p][c] = f2bf(fv);
      gB[p][c] = f2bf(gv);
    }
    int c = tid;
    for (int p2 = 0; p2 < 64; ++p2) {
      int py2 = p2 >> 3, px2 = p2 & 7;
      int gy2 = wy * 4 + py2, gx2 = wx * 4 + px2;
      float hv = 0.f;
      if (gy2 < Hdim && gx2 < Wdim)
        hv = Hh[(((size_t)b * C + c) * Hdim + gy2) * Wdim + gx2];
      hA[c][p2] = f2bf(hv);
    }
  }
  __syncthreads();

  const int lane = tid & 31, wave = tid >> 5;
  const int half = lane >> 4, l15 = lane & 15;
  const int kbA = half * 8, ksB = half * 16;

  for (int ti = wave; ti < 16; ti += 8) {
    int qt = ti >> 2, kt = ti & 3;
    v16bf a, bb;
    #pragma unroll
    for (int r = 0; r < 8; ++r) {
      int kidx = kbA + ((r >= 4) ? 16 : 0) + 2 * (r & 3);
      a[2 * r]     = fA[qt * 16 + l15][kidx];
      a[2 * r + 1] = fA[qt * 16 + l15][kidx + 1];
    }
    #pragma unroll
    for (int e = 0; e < 16; ++e) bb[e] = gB[kt * 16 + l15][kbA + (e & 7) + ((e >> 3) << 4)];
    v8f s = {0.f,0.f,0.f,0.f,0.f,0.f,0.f,0.f};
    s = __builtin_amdgcn_wmma_f32_16x16x32_bf16(false, a, false, bb,
                                                (short)0, s, false, false);
    #pragma unroll
    for (int r = 0; r < 8; ++r)
      sc[qt * 16 + r + half * 8][kt * 16 + l15] = s[r];
  }
  __syncthreads();

  if (tid < 64) {
    int q = tid;
    float mx = -1e38f;
    for (int k = 0; k < 64; ++k) {
      float v = sc[q][k] + kmask[k];
      sc[q][k] = v;
      mx = fmaxf(mx, v);
    }
    float sum = 0.f;
    for (int k = 0; k < 64; ++k) {
      float e = __expf(sc[q][k] - mx);
      sc[q][k] = e; sum += e;
    }
    float inv = 1.f / sum;
    for (int k = 0; k < 64; ++k) aB[q][k] = f2bf(sc[q][k] * inv);
  }
  __syncthreads();

  for (int ti = wave; ti < 64; ti += 8) {
    int ct = ti >> 2, qt = ti & 3;
    v8f o = {0.f,0.f,0.f,0.f,0.f,0.f,0.f,0.f};
    #pragma unroll
    for (int ks = 0; ks < 2; ++ks) {
      v16bf a, bb;
      #pragma unroll
      for (int r = 0; r < 8; ++r) {
        int kidx = ks * 32 + kbA + ((r >= 4) ? 16 : 0) + 2 * (r & 3);
        a[2 * r]     = hA[ct * 16 + l15][kidx];
        a[2 * r + 1] = hA[ct * 16 + l15][kidx + 1];
      }
      #pragma unroll
      for (int e = 0; e < 16; ++e) bb[e] = aB[qt * 16 + l15][ks * 32 + kbA + (e & 7) + ((e >> 3) << 4)];
      o = __builtin_amdgcn_wmma_f32_16x16x32_bf16(false, a, false, bb,
                                                  (short)0, o, false, false);
    }
    #pragma unroll
    for (int r = 0; r < 8; ++r) {
      int c = ct * 16 + r + half * 8;
      int q = qt * 16 + l15;
      int py = q >> 3, px = q & 7;
      if (py < 4 && px < 4) {
        int oy = wy * 4 + py, ox = wx * 4 + px;
        out[(((size_t)b * C + c) * Hdim + oy) * Wdim + ox] = o[r];
      }
    }
  }
}

// ---------------------------------------------------------------------------
// Host-side orchestration
// ---------------------------------------------------------------------------
static inline int ceil_div(int a, int b) { return (a + b - 1) / b; }

static void launch_conv(hipStream_t stream, bf16* wp, const float* zpad,
                        const float* X, const float* Wdev, const float* bias,
                        float* Y, int N, int Cin, int H, int W, int Cout,
                        int kh, int kw, int stride, int pad, int mode, int act) {
  int Hout, Wout;
  if (mode == 0) {
    Hout = (H + 2 * pad - kh) / stride + 1;
    Wout = (W + 2 * pad - kw) / stride + 1;
  } else {
    Hout = 2 * H; Wout = 2 * W;
  }
  int Kdim = Cin * kh * kw;
  int Kpad = ceil_div(Kdim, 32) * 32;
  int Mpad = ceil_div(Cout, 64) * 64;
  int Npos = N * Hout * Wout;
  pack_weights<<<ceil_div(Mpad * Kpad, 256), 256, 0, stream>>>(
      Wdev, wp, Cout, Cin, kh, kw, Kdim, Mpad, Kpad, mode);
  dim3 grid(ceil_div(Npos, BN), Mpad / BM);
  conv_gemm<<<grid, 256, 0, stream>>>(wp, X, bias, zpad, Y, N, Cin, H, W, Cout,
                                      Hout, Wout, kh, kw, stride, pad, mode, act,
                                      Kdim, Kpad, Npos);
}

static void run_adalin(hipStream_t stream, float* x, float* st, float* lst,
                       const float* rho, const float* gam, const float* bet,
                       int N, int C, int HW) {
  stats_kernel<<<N * C, 256, 0, stream>>>(x, st, HW);
  layer_reduce<<<ceil_div(C, 64), 64, 0, stream>>>(st, lst, N, C, HW);
  size_t total = (size_t)N * C * HW;
  adalin_apply<<<4096, 256, 0, stream>>>(x, st, lst, rho, gam, bet, C, HW, total, 1);
}

static void run_inorm(hipStream_t stream, float* st, const float* xin, float* xout,
                      const float* res, int N, int C, int HW, int relu) {
  stats_kernel<<<N * C, 256, 0, stream>>>(xin, st, HW);
  size_t total = (size_t)N * C * HW;
  inorm_apply<<<4096, 256, 0, stream>>>(xin, xout, st, res, HW, total, relu);
}

extern "C" void kernel_launch(void* const* d_in, const int* in_sizes, int n_in,
                              void* d_out, int out_size, void* d_ws, size_t ws_size,
                              hipStream_t stream) {
  (void)in_sizes; (void)n_in; (void)out_size; (void)ws_size;
  const float* x    = (const float*)d_in[0];
  const float* w_e1 = (const float*)d_in[1];  const float* b_e1 = (const float*)d_in[2];
  const float* rho1 = (const float*)d_in[3];  const float* gam1 = (const float*)d_in[4];
  const float* bet1 = (const float*)d_in[5];
  const float* w_e2 = (const float*)d_in[6];  const float* b_e2 = (const float*)d_in[7];
  const float* rho2 = (const float*)d_in[8];  const float* gam2 = (const float*)d_in[9];
  const float* bet2 = (const float*)d_in[10];
  const float* w_e3 = (const float*)d_in[11]; const float* b_e3 = (const float*)d_in[12];
  const float* rho3 = (const float*)d_in[13]; const float* gam3 = (const float*)d_in[14];
  const float* bet3 = (const float*)d_in[15];
  const float* rw1  = (const float*)d_in[16]; const float* rb1  = (const float*)d_in[17];
  const float* rw2  = (const float*)d_in[18]; const float* rb2  = (const float*)d_in[19];
  const float* w_f  = (const float*)d_in[20]; const float* b_f  = (const float*)d_in[21];
  const float* w_g  = (const float*)d_in[22]; const float* b_g  = (const float*)d_in[23];
  const float* w_h  = (const float*)d_in[24]; const float* b_h  = (const float*)d_in[25];
  const float* w_d1 = (const float*)d_in[26]; const float* b_d1 = (const float*)d_in[27];
  const float* rho4 = (const float*)d_in[28]; const float* gam4 = (const float*)d_in[29];
  const float* bet4 = (const float*)d_in[30];
  const float* w_d2 = (const float*)d_in[31]; const float* b_d2 = (const float*)d_in[32];
  const float* rho5 = (const float*)d_in[33]; const float* gam5 = (const float*)d_in[34];
  const float* bet5 = (const float*)d_in[35];
  const float* w_d3 = (const float*)d_in[36]; const float* b_d3 = (const float*)d_in[37];

  float* ws = (float*)d_ws;
  size_t o = 0;
  float* Abuf = ws + o; o += 67108864;   // 4x64x512x512  (e1 out, later d2 out)
  float* Bbuf = ws + o; o += 33554432;   // 4x128x256x256 (e2 out, later d1 out)
  float* C0   = ws + o; o += 16777216;   // 4x256x128x128 (residual stream)
  float* C1   = ws + o; o += 16777216;   // conv scratch
  float* C2   = ws + o; o += 16777216;   // conv scratch / h
  float* Fb   = ws + o; o += 2097152;    // 4x32x128x128
  float* Gb   = ws + o; o += 2097152;
  float* st   = ws + o; o += 4096;
  float* lst  = ws + o; o += 1024;
  float* zpad = ws + o; o += 256;        // zero landing pad for async gathers
  bf16*  wp   = (bf16*)(ws + o);

  float* img  = (float*)d_out;            // 4x3x512x512
  float* attn = (float*)d_out + 3145728;  // 4x256x128x128

  zero_fill<<<1, 256, 0, stream>>>(zpad, 256);

  // ---- encoder ----
  launch_conv(stream, wp, zpad, x, w_e1, b_e1, Abuf, 4, 3, 512, 512, 64, 7, 7, 1, 3, 0, 0);
  run_adalin(stream, Abuf, st, lst, rho1, gam1, bet1, 4, 64, 512 * 512);
  launch_conv(stream, wp, zpad, Abuf, w_e2, b_e2, Bbuf, 4, 64, 512, 512, 128, 3, 3, 2, 1, 0, 0);
  run_adalin(stream, Bbuf, st, lst, rho2, gam2, bet2, 4, 128, 256 * 256);
  launch_conv(stream, wp, zpad, Bbuf, w_e3, b_e3, C0, 4, 128, 256, 256, 256, 3, 3, 2, 1, 0, 0);
  run_adalin(stream, C0, st, lst, rho3, gam3, bet3, 4, 256, 128 * 128);

  // ---- residual blocks ----
  for (int i = 0; i < 4; ++i) {
    const float* w1 = rw1 + (size_t)i * 256 * 256 * 9;
    const float* bb1 = rb1 + i * 256;
    const float* w2 = rw2 + (size_t)i * 256 * 256 * 9;
    const float* bb2 = rb2 + i * 256;
    launch_conv(stream, wp, zpad, C0, w1, bb1, C1, 4, 256, 128, 128, 256, 3, 3, 1, 1, 0, 0);
    run_inorm(stream, st, C1, C1, nullptr, 4, 256, 128 * 128, 1);
    launch_conv(stream, wp, zpad, C1, w2, bb2, C2, 4, 256, 128, 128, 256, 3, 3, 1, 1, 0, 0);
    run_inorm(stream, st, C2, C0, C0, 4, 256, 128 * 128, 1);
  }

  // ---- windowed attention (second output) ----
  launch_conv(stream, wp, zpad, C0, w_f, b_f, Fb, 4, 256, 128, 128, 32, 1, 1, 1, 0, 0, 0);
  launch_conv(stream, wp, zpad, C0, w_g, b_g, Gb, 4, 256, 128, 128, 32, 1, 1, 1, 0, 0, 0);
  launch_conv(stream, wp, zpad, C0, w_h, b_h, C2, 4, 256, 128, 128, 256, 1, 1, 1, 0, 0, 0);
  win_attn<<<4 * 32 * 32, 256, 0, stream>>>(Fb, Gb, C2, attn,
                                            4, 256, 32, 128, 128, 32, 32);

  // ---- decoder ----
  launch_conv(stream, wp, zpad, C0, w_d1, b_d1, Bbuf, 4, 256, 128, 128, 128, 3, 3, 2, 1, 1, 0);
  run_adalin(stream, Bbuf, st, lst, rho4, gam4, bet4, 4, 128, 256 * 256);
  launch_conv(stream, wp, zpad, Bbuf, w_d2, b_d2, Abuf, 4, 128, 256, 256, 64, 3, 3, 2, 1, 1, 0);
  run_adalin(stream, Abuf, st, lst, rho5, gam5, bet5, 4, 64, 512 * 512);
  launch_conv(stream, wp, zpad, Abuf, w_d3, b_d3, img, 4, 64, 512, 512, 3, 7, 7, 1, 3, 0, 1);
}
